// Atom_GTN_41532333752496
// MI455X (gfx1250) — compile-verified, matrix operand order checked by reading, not run on hardware
//
#include <hip/hip_runtime.h>
#include <math.h>

#define NNODE 50000
#define NEDGE 800000
#define FIN   128
#define OUTD  64
#define HEADS 4
#define HD    256          // HEADS*OUTD
#define SCALEF 0.125f      // 1/sqrt(64)

typedef __bf16 bf16_t;
typedef __attribute__((ext_vector_type(16))) __bf16 v16bf;
typedef __attribute__((ext_vector_type(8)))  float  v8f;

union Frag { uint4 u[2]; v16bf v; };

// ---------------- helpers ----------------
__device__ __forceinline__ unsigned short f2bf(float f) {
    unsigned u = __float_as_uint(f);
    u += 0x7fffu + ((u >> 16) & 1u);          // round-to-nearest-even
    return (unsigned short)(u >> 16);
}
__device__ __forceinline__ unsigned fkey(float f) {      // order-preserving map
    unsigned u = __float_as_uint(f);
    return ((int)u < 0) ? ~u : (u | 0x80000000u);
}
__device__ __forceinline__ float funkey(unsigned k) {
    unsigned u = ((int)k < 0) ? (k & 0x7fffffffu) : ~k;
    return __uint_as_float(u);
}

// ---------------- f32 -> bf16 convert ----------------
__global__ void k_cvt_bf16(const float* __restrict__ in, unsigned short* __restrict__ out, int n) {
    int i = blockIdx.x * blockDim.x + threadIdx.x;
    if (i < n) out[i] = f2bf(in[i]);
}

// ---------------- transpose + convert: W[K x Ncol] f32 -> WT[Ncol x K] bf16 ----------------
__global__ void k_transpose_cvt(const float* __restrict__ W, unsigned short* __restrict__ WT,
                                int K, int Ncol) {
    int i = blockIdx.x * blockDim.x + threadIdx.x;
    if (i >= K * Ncol) return;
    int k = i / Ncol, c = i % Ncol;
    WT[(size_t)c * K + k] = f2bf(W[i]);
}

// ---------------- bf16 WMMA GEMM: C[M x Ncol] = A[M x K] * BT[Ncol x K]^T + bias ----------------
// block = 128 threads (4 waves); wave w handles colTile = blockIdx.y*4 + w, rowTile = blockIdx.x
__global__ void k_gemm_wmma(const unsigned short* __restrict__ A,
                            const unsigned short* __restrict__ BT,
                            const float* __restrict__ bias,
                            float* __restrict__ C,
                            int M, int K, int Ncol) {
    const int lane = threadIdx.x & 31;
    const int wave = threadIdx.x >> 5;
    const int rowTile = blockIdx.x;
    const int colTile = blockIdx.y * 4 + wave;
    const int m = rowTile * 16 + (lane & 15);
    const int n = colTile * 16 + (lane & 15);
    const int koff = (lane >> 4) * 8;         // lanes 16-31 start at K=8 (bf16 A/B layout)

    const unsigned short* arow = A  + (size_t)m * K;
    const unsigned short* brow = BT + (size_t)n * K;

    v8f acc = {0.f, 0.f, 0.f, 0.f, 0.f, 0.f, 0.f, 0.f};
    for (int kb = 0; kb < K; kb += 32) {
        Frag fa, fb;
        fa.u[0] = *(const uint4*)(arow + kb + koff);        // K = kb+koff .. +7
        fa.u[1] = *(const uint4*)(arow + kb + koff + 16);   // K = kb+koff+16 .. +23
        fb.u[0] = *(const uint4*)(brow + kb + koff);
        fb.u[1] = *(const uint4*)(brow + kb + koff + 16);
        __builtin_prefetch(arow + kb + 32, 0, 0);
        acc = __builtin_amdgcn_wmma_f32_16x16x32_bf16(
                  false, fa.v, false, fb.v, (short)0, acc, false, false);
    }
    const int rbase = rowTile * 16 + ((lane >> 4) ? 8 : 0);
    const float bcol = bias ? bias[n] : 0.f;
    #pragma unroll
    for (int v = 0; v < 8; ++v)
        C[(size_t)(rbase + v) * Ncol + n] = acc[v] + bcol;
}

// ---------------- edge pass A: logits + segment max ----------------
// one wave per edge; We(16x256 f32) staged in LDS
__global__ void k_edge_logits(const int* __restrict__ srcI, const int* __restrict__ dstI,
                              const float* __restrict__ ea, const float* __restrict__ We,
                              const float* __restrict__ Q, const float* __restrict__ Kf,
                              float* __restrict__ logitBuf, unsigned* __restrict__ mKey, int E) {
    __shared__ float weS[16 * HD];
    for (int i = threadIdx.x; i < 16 * HD; i += blockDim.x) weS[i] = We[i];
    __syncthreads();
    const int lane = threadIdx.x & 31;
    const int wave = threadIdx.x >> 5;
    const int c0 = lane * 8;
    for (int e = blockIdx.x * 8 + wave; e < E; e += gridDim.x * 8) {
        const int sI = srcI[e], dI = dstI[e];
        float myea = (lane < 16) ? ea[(size_t)e * 16 + lane] : 0.f;
        float ev[8];
        #pragma unroll
        for (int u = 0; u < 8; ++u) ev[u] = 0.f;
        #pragma unroll
        for (int j = 0; j < 16; ++j) {
            float aj = __shfl(myea, j, 32);
            #pragma unroll
            for (int u = 0; u < 8; ++u) ev[u] += aj * weS[j * HD + c0 + u];
        }
        const float* qr = Q  + (size_t)dI * HD + c0;
        const float* kr = Kf + (size_t)sI * HD + c0;
        float acc = 0.f;
        #pragma unroll
        for (int u = 0; u < 8; ++u) acc += qr[u] * (kr[u] + ev[u]);
        acc += __shfl_xor(acc, 4, 32);
        acc += __shfl_xor(acc, 2, 32);
        acc += __shfl_xor(acc, 1, 32);
        if ((lane & 7) == 0) {
            int h = lane >> 3;
            float lg = acc * SCALEF;
            logitBuf[(size_t)e * HEADS + h] = lg;
            atomicMax(&mKey[dI * HEADS + h], fkey(lg));
        }
    }
}

// ---------------- edge pass B: exp(logit - m), segment sum ----------------
__global__ void k_edge_exp(const int* __restrict__ dstI, const unsigned* __restrict__ mKey,
                           float* __restrict__ logitBuf, float* __restrict__ sBuf, int E4) {
    int i = blockIdx.x * blockDim.x + threadIdx.x;
    if (i >= E4) return;
    int e = i >> 2, h = i & 3;
    int dI = dstI[e];
    float m = funkey(mKey[dI * HEADS + h]);
    float a = __expf(logitBuf[i] - m);
    logitBuf[i] = a;
    atomicAdd(&sBuf[dI * HEADS + h], a);
}

// ---------------- edge pass C: weighted scatter aggregation ----------------
__global__ void k_edge_agg(const int* __restrict__ srcI, const int* __restrict__ dstI,
                           const float* __restrict__ ea, const float* __restrict__ We,
                           const float* __restrict__ V, const float* __restrict__ logitBuf,
                           const float* __restrict__ sBuf, float* __restrict__ agg, int E) {
    __shared__ float weS[16 * HD];
    for (int i = threadIdx.x; i < 16 * HD; i += blockDim.x) weS[i] = We[i];
    __syncthreads();
    const int lane = threadIdx.x & 31;
    const int wave = threadIdx.x >> 5;
    const int c0 = lane * 8;
    const int h = lane >> 3;
    for (int e = blockIdx.x * 8 + wave; e < E; e += gridDim.x * 8) {
        const int sI = srcI[e], dI = dstI[e];
        float alpha = logitBuf[(size_t)e * HEADS + h] / sBuf[dI * HEADS + h];
        float myea = (lane < 16) ? ea[(size_t)e * 16 + lane] : 0.f;
        float ev[8];
        #pragma unroll
        for (int u = 0; u < 8; ++u) ev[u] = 0.f;
        #pragma unroll
        for (int j = 0; j < 16; ++j) {
            float aj = __shfl(myea, j, 32);
            #pragma unroll
            for (int u = 0; u < 8; ++u) ev[u] += aj * weS[j * HD + c0 + u];
        }
        const float* vr = V + (size_t)sI * HD + c0;
        float* ar = agg + (size_t)dI * HD + c0;
        #pragma unroll
        for (int u = 0; u < 8; ++u)
            atomicAdd(&ar[u], (vr[u] + ev[u]) * alpha);
    }
}

// ---------------- node finalize: head mean + skip, LayerNorm(64), exact GELU ----------------
__global__ void k_finalize(const float* __restrict__ agg, const float* __restrict__ S0,
                           const float* __restrict__ g, const float* __restrict__ b,
                           float* __restrict__ out) {
    const int n = blockIdx.x, d = threadIdx.x;   // blockDim = 64
    const float* a = agg + (size_t)n * HD;
    float v = 0.25f * (a[d] + a[64 + d] + a[128 + d] + a[192 + d]) + S0[(size_t)n * OUTD + d];
    __shared__ float sh[64];
    sh[d] = v; __syncthreads();
    for (int off = 32; off >= 1; off >>= 1) { if (d < off) sh[d] += sh[d + off]; __syncthreads(); }
    float mu = sh[0] * (1.f / 64.f); __syncthreads();
    float dv = v - mu;
    sh[d] = dv * dv; __syncthreads();
    for (int off = 32; off >= 1; off >>= 1) { if (d < off) sh[d] += sh[d + off]; __syncthreads(); }
    float var = sh[0] * (1.f / 64.f);
    float y = dv * rsqrtf(var + 1e-5f) * g[d] + b[d];
    out[(size_t)n * OUTD + d] = 0.5f * y * (1.f + erff(y * 0.70710678118654752f));
}

// ---------------- host orchestration ----------------
extern "C" void kernel_launch(void* const* d_in, const int* in_sizes, int n_in,
                              void* d_out, int out_size, void* d_ws, size_t ws_size,
                              hipStream_t stream) {
    (void)in_sizes; (void)n_in; (void)out_size; (void)ws_size;
    const float* x    = (const float*)d_in[0];
    const int*   ei   = (const int*)  d_in[1];
    const float* ea   = (const float*)d_in[2];
    const float* ln_g = (const float*)d_in[30];
    const float* ln_b = (const float*)d_in[31];
    const int* srcI = ei;
    const int* dstI = ei + NEDGE;

    // workspace carve (256B aligned)
    char* ws = (char*)d_ws; size_t off = 0;
    auto carve = [&](size_t bytes) -> void* {
        void* p = ws + off; off += (bytes + 255) & ~(size_t)255; return p;
    };
    unsigned short* hb   = (unsigned short*)carve((size_t)NNODE * FIN * 2);
    unsigned short* WTq  = (unsigned short*)carve((size_t)HD * FIN * 2);
    unsigned short* WTk  = (unsigned short*)carve((size_t)HD * FIN * 2);
    unsigned short* WTv  = (unsigned short*)carve((size_t)HD * FIN * 2);
    unsigned short* WTs  = (unsigned short*)carve((size_t)OUTD * FIN * 2);
    float* Qb   = (float*)carve((size_t)NNODE * HD * 4);
    float* Kb   = (float*)carve((size_t)NNODE * HD * 4);
    float* Vb   = (float*)carve((size_t)NNODE * HD * 4);
    float* S0   = (float*)carve((size_t)NNODE * OUTD * 4);
    float* hbuf = (float*)carve((size_t)NNODE * OUTD * 4);
    float* lgb  = (float*)carve((size_t)NEDGE * HEADS * 4);
    unsigned* mKey = (unsigned*)carve((size_t)NNODE * HEADS * 4);
    float* sBuf = (float*)carve((size_t)NNODE * HEADS * 4);
    float* agg  = (float*)carve((size_t)NNODE * HD * 4);

    for (int l = 0; l < 3; ++l) {
        const int fin = (l == 0) ? FIN : OUTD;
        const float* hin = (l == 0) ? x : hbuf;
        void* const* p = d_in + 3 + 9 * l;
        const float *Wq = (const float*)p[0], *bq = (const float*)p[1];
        const float *Wk = (const float*)p[2], *bk = (const float*)p[3];
        const float *Wv = (const float*)p[4], *bv = (const float*)p[5];
        const float *We = (const float*)p[6];
        const float *Ws = (const float*)p[7], *bs = (const float*)p[8];

        // convert activations + weights to bf16 (transposed weights for contiguous B fragments)
        int nh = NNODE * fin;
        k_cvt_bf16<<<(nh + 255) / 256, 256, 0, stream>>>(hin, hb, nh);
        k_transpose_cvt<<<(fin * HD + 255) / 256, 256, 0, stream>>>(Wq, WTq, fin, HD);
        k_transpose_cvt<<<(fin * HD + 255) / 256, 256, 0, stream>>>(Wk, WTk, fin, HD);
        k_transpose_cvt<<<(fin * HD + 255) / 256, 256, 0, stream>>>(Wv, WTv, fin, HD);
        k_transpose_cvt<<<(fin * OUTD + 255) / 256, 256, 0, stream>>>(Ws, WTs, fin, OUTD);

        // node GEMMs via WMMA (M = 50000 = 3125*16 exactly)
        dim3 gg(NNODE / 16, HD / 64);
        k_gemm_wmma<<<gg, 128, 0, stream>>>(hb, WTq, bq, Qb, NNODE, fin, HD);
        k_gemm_wmma<<<gg, 128, 0, stream>>>(hb, WTk, bk, Kb, NNODE, fin, HD);
        k_gemm_wmma<<<gg, 128, 0, stream>>>(hb, WTv, bv, Vb, NNODE, fin, HD);
        dim3 gs(NNODE / 16, 1);
        k_gemm_wmma<<<gs, 128, 0, stream>>>(hb, WTs, bs, S0, NNODE, fin, OUTD);

        // zero-init segment buffers (fkey(-inf-ish) == 0 for the max)
        hipMemsetAsync(mKey, 0, (size_t)NNODE * HEADS * 4, stream);
        hipMemsetAsync(sBuf, 0, (size_t)NNODE * HEADS * 4, stream);
        hipMemsetAsync(agg,  0, (size_t)NNODE * HD * 4,    stream);

        // edge phase
        k_edge_logits<<<1024, 256, 0, stream>>>(srcI, dstI, ea, We, Qb, Kb, lgb, mKey, NEDGE);
        k_edge_exp<<<(NEDGE * HEADS + 255) / 256, 256, 0, stream>>>(dstI, mKey, lgb, sBuf, NEDGE * HEADS);
        k_edge_agg<<<1024, 256, 0, stream>>>(srcI, dstI, ea, We, Vb, lgb, sBuf, agg, NEDGE);

        // finalize
        float* outp = (l == 2) ? (float*)d_out : hbuf;
        k_finalize<<<NNODE, 64, 0, stream>>>(agg, S0, ln_g, ln_b, outp);
    }
}
